// SemanticAlignmentModule_26757646254082
// MI455X (gfx1250) — matrix operand order repaired and checked by reading, not compiled, not run on hardware
//
#include <hip/hip_runtime.h>
#include <hip/hip_bf16.h>

// ---------- types ----------
typedef __bf16 bf16_t;
typedef bf16_t v16bf __attribute__((ext_vector_type(16)));
typedef float  v8f   __attribute__((ext_vector_type(8)));
typedef unsigned int u32x4 __attribute__((ext_vector_type(4)));

#define LN_EPS   1e-5f
#define INV_TEMP (1.0f / 0.07f)

__device__ __forceinline__ unsigned short f2bf(float f) {
  unsigned int u = __float_as_uint(f);
  u += 0x7FFFu + ((u >> 16) & 1u);           // round-to-nearest-even
  return (unsigned short)(u >> 16);
}
__device__ __forceinline__ float bf2f(unsigned short h) {
  return __uint_as_float(((unsigned int)h) << 16);
}
__device__ __forceinline__ float gelu_exact(float x) {
  return 0.5f * x * (1.0f + erff(x * 0.70710678118654752f));
}

union AFrag  { v16bf v; unsigned short s[16]; };
union BFragU { v16bf v; u32x4 q[2]; };

// ---------- elementwise f32 -> bf16 ----------
__global__ void f32_to_bf16_kernel(const float* __restrict__ s,
                                   unsigned short* __restrict__ d, int n) {
  for (int i = blockIdx.x * blockDim.x + threadIdx.x; i < n;
       i += gridDim.x * blockDim.x)
    d[i] = f2bf(s[i]);
}

// ---------- text prep: raw bf16 + l2-normalized bf16 (row = 768) ----------
__global__ __launch_bounds__(256) void prep_text_kernel(
    const float* __restrict__ text, unsigned short* __restrict__ textB,
    unsigned short* __restrict__ keytB) {
  __shared__ float red[256];
  const int tid = threadIdx.x;
  const long long base = (long long)blockIdx.x * 768;
  const float* tr = text + base;
  float a = tr[tid], b = tr[tid + 256], c = tr[tid + 512];
  textB[base + tid]       = f2bf(a);
  textB[base + tid + 256] = f2bf(b);
  textB[base + tid + 512] = f2bf(c);
  red[tid] = a * a + b * b + c * c;
  __syncthreads();
  for (int off = 128; off > 0; off >>= 1) {
    if (tid < off) red[tid] += red[tid + off];
    __syncthreads();
  }
  float inv = 1.0f / fmaxf(sqrtf(red[0]), 1e-12f);
  keytB[base + tid]       = f2bf(a * inv);
  keytB[base + tid + 256] = f2bf(b * inv);
  keytB[base + tid + 512] = f2bf(c * inv);
}

// ---------- LayerNorm over 512 (one block per row) ----------
__global__ __launch_bounds__(256) void ln512_kernel(
    const float* x, const float* __restrict__ g, const float* __restrict__ be,
    float* outF, unsigned short* outB) {
  __shared__ float red[256];
  const int tid = threadIdx.x;
  const long long row = blockIdx.x;
  const float* xr = x + row * 512;
  float v0 = xr[tid], v1 = xr[tid + 256];
  red[tid] = v0 + v1;
  __syncthreads();
  for (int off = 128; off > 0; off >>= 1) {
    if (tid < off) red[tid] += red[tid + off];
    __syncthreads();
  }
  float mu = red[0] * (1.0f / 512.0f);
  __syncthreads();
  float d0 = v0 - mu, d1 = v1 - mu;
  red[tid] = d0 * d0 + d1 * d1;
  __syncthreads();
  for (int off = 128; off > 0; off >>= 1) {
    if (tid < off) red[tid] += red[tid + off];
    __syncthreads();
  }
  float rstd = rsqrtf(red[0] * (1.0f / 512.0f) + LN_EPS);
  float y0 = d0 * rstd * g[tid] + be[tid];
  float y1 = d1 * rstd * g[tid + 256] + be[tid + 256];
  if (outF) {
    outF[row * 512 + tid] = y0;
    outF[row * 512 + tid + 256] = y1;
  }
  if (outB) {
    outB[row * 512 + tid] = f2bf(y0);
    outB[row * 512 + tid + 256] = f2bf(y1);
  }
}

// ---------- l2-normalize bf16 rows of 768 in place ----------
__global__ __launch_bounds__(256) void l2norm768_inplace_kernel(
    unsigned short* qb) {
  __shared__ float red[256];
  const int tid = threadIdx.x;
  unsigned short* r = qb + (long long)blockIdx.x * 768;
  float a = bf2f(r[tid]), b = bf2f(r[tid + 256]), c = bf2f(r[tid + 512]);
  red[tid] = a * a + b * b + c * c;
  __syncthreads();
  for (int off = 128; off > 0; off >>= 1) {
    if (tid < off) red[tid] += red[tid + off];
    __syncthreads();
  }
  float inv = 1.0f / fmaxf(sqrtf(red[0]), 1e-12f);
  r[tid]       = f2bf(a * inv);
  r[tid + 256] = f2bf(b * inv);
  r[tid + 512] = f2bf(c * inv);
}

// ---------- generic bf16 WMMA GEMM: C[M,N] = A[M,K] * B[K,N] (+bias, epilogue)
// block = 256 threads (8 waves), tile 128(M) x 128(N), K-step 32,
// double-buffered LDS. Wave tile 32x64 -> 8 WMMA per K-step per wave.
// mode 0: outBF = bf16(acc+bias)
// mode 1: outBF = bf16(gelu(acc+bias))
// mode 2: outF32 = acc + bias + addF32 (fp32)
__global__ __launch_bounds__(256) void gemm_bf16_kernel(
    const unsigned short* __restrict__ A, long long sAb, int lda,
    const unsigned short* __restrict__ Bw, long long sBb,
    int N, int Kd, const float* __restrict__ bias,
    unsigned short* outBF, float* outF32, const float* addF32,
    long long sOb, int mode) {
  __shared__ __align__(16) unsigned short As[2][128][32];
  __shared__ __align__(16) unsigned short Bs[2][32][128];
  const int bz = blockIdx.z;
  const unsigned short* Ab = A + (long long)bz * sAb;
  const unsigned short* Bb = Bw + (long long)bz * sBb;
  const long long obase = (long long)bz * sOb;
  const int tid = threadIdx.x;
  const int lane = tid & 31;
  const int wid = tid >> 5;
  const int wy = wid & 3;   // 4 waves along M (32 rows each)
  const int wx = wid >> 2;  // 2 waves along N (64 cols each)
  const int m0 = blockIdx.y * 128;
  const int n0 = blockIdx.x * 128;

  // per-thread staging coordinates (2 x 16B vectors for A, 2 for B)
  const int ar0 = (tid * 2) >> 2, ag0 = (tid * 2) & 3;
  const int ar1 = (tid * 2 + 1) >> 2, ag1 = (tid * 2 + 1) & 3;
  const int br0 = (tid * 2) >> 4, bg0 = (tid * 2) & 15;
  const int br1 = (tid * 2 + 1) >> 4, bg1 = (tid * 2 + 1) & 15;
  const unsigned short* aSrc0 = Ab + (long long)(m0 + ar0) * lda + ag0 * 8;
  const unsigned short* aSrc1 = Ab + (long long)(m0 + ar1) * lda + ag1 * 8;
  const unsigned short* bSrc0 = Bb + (long long)br0 * N + n0 + bg0 * 8;
  const unsigned short* bSrc1 = Bb + (long long)br1 * N + n0 + bg1 * 8;

  v8f acc[2][4] = {};
  u32x4 aReg0, aReg1, bReg0, bReg1;

  // prologue: stage tile 0 into buffer 0
  aReg0 = *(const u32x4*)aSrc0;
  aReg1 = *(const u32x4*)aSrc1;
  bReg0 = *(const u32x4*)bSrc0;
  bReg1 = *(const u32x4*)bSrc1;
  *(u32x4*)&As[0][ar0][ag0 * 8] = aReg0;
  *(u32x4*)&As[0][ar1][ag1 * 8] = aReg1;
  *(u32x4*)&Bs[0][br0][bg0 * 8] = bReg0;
  *(u32x4*)&Bs[0][br1][bg1 * 8] = bReg1;
  __syncthreads();

  int buf = 0;
  for (int kk = 0; kk < Kd; kk += 32) {
    const int nxt = buf ^ 1;
    const bool more = (kk + 32) < Kd;
    if (more) {  // fetch next tile into registers (overlaps with WMMA below)
      aReg0 = *(const u32x4*)(aSrc0 + kk + 32);
      aReg1 = *(const u32x4*)(aSrc1 + kk + 32);
      bReg0 = *(const u32x4*)(bSrc0 + (long long)(kk + 32) * N);
      bReg1 = *(const u32x4*)(bSrc1 + (long long)(kk + 32) * N);
    }

    AFrag a[2];
#pragma unroll
    for (int ti = 0; ti < 2; ++ti) {
      int m = wy * 32 + ti * 16 + (lane & 15);
      int h = (lane >> 4) * 8;
#pragma unroll
      for (int p = 0; p < 16; ++p) {
        int k = (p & 7) + h + ((p >> 3) << 4);  // ISA 16-bit A 16x32 layout
        a[ti].s[p] = As[buf][m][k];
      }
    }
    BFragU bfr[4];
#pragma unroll
    for (int tj = 0; tj < 4; ++tj) {
      int nb = wx * 64 + tj * 16 + (lane & 15);  // lane = N col, elems = K
      int h = (lane >> 4) * 8;
#pragma unroll
      for (int p = 0; p < 16; ++p) {
        int k = (p & 7) + h + ((p >> 3) << 4);
        ((unsigned short*)&bfr[tj])[p] = Bs[buf][k][nb];
      }
    }
#pragma unroll
    for (int ti = 0; ti < 2; ++ti)
#pragma unroll
      for (int tj = 0; tj < 4; ++tj)
        acc[ti][tj] = __builtin_amdgcn_wmma_f32_16x16x32_bf16(
            false, a[ti].v, false, bfr[tj].v, (short)0, acc[ti][tj], false,
            false);

    if (more) {  // write next tile (prev readers of nxt passed last barrier)
      *(u32x4*)&As[nxt][ar0][ag0 * 8] = aReg0;
      *(u32x4*)&As[nxt][ar1][ag1 * 8] = aReg1;
      *(u32x4*)&Bs[nxt][br0][bg0 * 8] = bReg0;
      *(u32x4*)&Bs[nxt][br1][bg1 * 8] = bReg1;
    }
    __syncthreads();
    buf = nxt;
  }

#pragma unroll
  for (int ti = 0; ti < 2; ++ti) {
#pragma unroll
    for (int tj = 0; tj < 4; ++tj) {
      int gn = n0 + wx * 64 + tj * 16 + (lane & 15);
      float bval = bias ? bias[gn] : 0.0f;
#pragma unroll
      for (int r = 0; r < 8; ++r) {
        int gm = m0 + wy * 32 + ti * 16 + r + ((lane >> 4) << 3);
        float v = acc[ti][tj][r] + bval;
        long long idx = obase + (long long)gm * N + gn;
        if (mode == 0)
          outBF[idx] = f2bf(v);
        else if (mode == 1)
          outBF[idx] = f2bf(gelu_exact(v));
        else
          outF32[idx] = v + (addF32 ? addF32[idx] : 0.0f);
      }
    }
  }
}

// ---------- fused sim (q·keytT) + top-5 mask + softmax -> bf16 attn ----------
// block = 256 threads (8 waves): 64 rows x 128 keys per workgroup, K=768.
__global__ __launch_bounds__(256) void sim_softmax_kernel(
    const unsigned short* __restrict__ qB,   // [B,4096,768]
    const unsigned short* __restrict__ ktB,  // [B,128,768] (l2-normalized)
    unsigned short* __restrict__ attnB) {    // [B,4096,128]
  __shared__ __align__(16) unsigned short Qs[64][32];
  __shared__ __align__(16) unsigned short Ks[32][128];
  __shared__ float Sim[64][128];
  const int b = blockIdx.y;
  const int row0 = blockIdx.x * 64;
  const unsigned short* q = qB + ((long long)b * 4096 + row0) * 768;
  const unsigned short* kt = ktB + (long long)b * 128 * 768;
  const int tid = threadIdx.x;
  const int lane = tid & 31;
  const int wid = tid >> 5;
  const int wy = wid & 1;   // 2 waves along rows
  const int wx = wid >> 1;  // 4 waves along keys

  v8f acc[2][2] = {};

  for (int kk = 0; kk < 768; kk += 32) {
    {  // Q tile 64x32
      int r = tid >> 2, g = tid & 3;
      *(u32x4*)&Qs[r][g * 8] =
          *(const u32x4*)&q[(long long)r * 768 + kk + g * 8];
    }
#pragma unroll
    for (int i = 0; i < 2; ++i) {  // K^T tile 32x128 (transpose on store)
      int id = tid + 256 * i;
      int n = id >> 2, g = id & 3;
      u32x4 v = *(const u32x4*)&kt[(long long)n * 768 + kk + g * 8];
      const unsigned short* ps = (const unsigned short*)&v;
#pragma unroll
      for (int j = 0; j < 8; ++j) Ks[g * 8 + j][n] = ps[j];
    }
    __syncthreads();

    AFrag a[2];
#pragma unroll
    for (int ti = 0; ti < 2; ++ti) {
      int m = wy * 32 + ti * 16 + (lane & 15);
      int h = (lane >> 4) * 8;
#pragma unroll
      for (int p = 0; p < 16; ++p) {
        int k = (p & 7) + h + ((p >> 3) << 4);
        a[ti].s[p] = Qs[m][k];
      }
    }
    BFragU bfr[2];
#pragma unroll
    for (int tj = 0; tj < 2; ++tj) {
      int nb = wx * 32 + tj * 16 + (lane & 15);
      int h = (lane >> 4) * 8;
#pragma unroll
      for (int p = 0; p < 16; ++p) {
        int k = (p & 7) + h + ((p >> 3) << 4);
        ((unsigned short*)&bfr[tj])[p] = Ks[k][nb];
      }
    }
#pragma unroll
    for (int ti = 0; ti < 2; ++ti)
#pragma unroll
      for (int tj = 0; tj < 2; ++tj)
        acc[ti][tj] = __builtin_amdgcn_wmma_f32_16x16x32_bf16(
            false, a[ti].v, false, bfr[tj].v, (short)0, acc[ti][tj], false,
            false);
    __syncthreads();
  }

#pragma unroll
  for (int ti = 0; ti < 2; ++ti)
#pragma unroll
    for (int tj = 0; tj < 2; ++tj)
#pragma unroll
      for (int r = 0; r < 8; ++r)
        Sim[wy * 32 + ti * 16 + r + ((lane >> 4) << 3)]
           [wx * 32 + tj * 16 + (lane & 15)] = acc[ti][tj][r];
  __syncthreads();

  if (tid < 64) {  // one thread per row: top-5 threshold + softmax/TEMP
    float t0 = -3.4e38f, t1 = t0, t2 = t0, t3 = t0, t4 = t0;
    for (int k2 = 0; k2 < 128; ++k2) {
      float s = Sim[tid][k2];
      if (s > t0) { t4 = t3; t3 = t2; t2 = t1; t1 = t0; t0 = s; }
      else if (s > t1) { t4 = t3; t3 = t2; t2 = t1; t1 = s; }
      else if (s > t2) { t4 = t3; t3 = t2; t2 = s; }
      else if (s > t3) { t4 = t3; t3 = s; }
      else if (s > t4) { t4 = s; }
    }
    float sum = 0.0f;
    for (int k2 = 0; k2 < 128; ++k2) {
      float s = Sim[tid][k2];
      if (s >= t4) sum += __expf((s - t0) * INV_TEMP);
    }
    float inv = 1.0f / sum;
    unsigned short* ar = attnB + ((long long)b * 4096 + row0 + tid) * 128;
    for (int k2 = 0; k2 < 128; ++k2) {
      float s = Sim[tid][k2];
      ar[k2] = (s >= t4) ? f2bf(__expf((s - t0) * INV_TEMP) * inv)
                         : (unsigned short)0;
    }
  }
}

// ---------- host ----------
extern "C" void kernel_launch(void* const* d_in, const int* in_sizes, int n_in,
                              void* d_out, int out_size, void* d_ws,
                              size_t ws_size, hipStream_t stream) {
  (void)in_sizes; (void)n_in; (void)out_size; (void)ws_size;
  const float* visual = (const float*)d_in[0];
  const float* text   = (const float*)d_in[1];
  const float* Wq  = (const float*)d_in[2];
  const float* bq  = (const float*)d_in[3];
  const float* Wv  = (const float*)d_in[4];
  const float* bvv = (const float*)d_in[5];
  const float* W1  = (const float*)d_in[6];
  const float* b1  = (const float*)d_in[7];
  const float* W2  = (const float*)d_in[8];
  const float* b2  = (const float*)d_in[9];
  const float* g1  = (const float*)d_in[10];
  const float* be1 = (const float*)d_in[11];
  const float* g2  = (const float*)d_in[12];
  const float* be2 = (const float*)d_in[13];
  float* out = (float*)d_out;

  const int Bb = 8, Ntok = 4096, C = 512, T = 768, Kk = 128, F = 2048;
  const long long MT = (long long)Bb * Ntok;  // 32768 rows

  char* cur = (char*)d_ws;
  auto alloc = [&](size_t bytes) {
    char* r = cur;
    cur += (bytes + 255) & ~(size_t)255;
    return r;
  };
  unsigned short* WqB    = (unsigned short*)alloc((size_t)C * T * 2);
  unsigned short* WvB    = (unsigned short*)alloc((size_t)T * C * 2);
  unsigned short* W1B    = (unsigned short*)alloc((size_t)C * F * 2);
  unsigned short* W2B    = (unsigned short*)alloc((size_t)F * C * 2);
  unsigned short* textB  = (unsigned short*)alloc((size_t)Bb * Kk * T * 2);
  unsigned short* keytB  = (unsigned short*)alloc((size_t)Bb * Kk * T * 2);
  unsigned short* valueB = (unsigned short*)alloc((size_t)Bb * Kk * C * 2);
  unsigned short* vislnB = (unsigned short*)alloc((size_t)MT * C * 2);
  unsigned short* qB     = (unsigned short*)alloc((size_t)MT * T * 2);
  unsigned short* attnB  = (unsigned short*)alloc((size_t)MT * Kk * 2);
  unsigned short* fusedB = (unsigned short*)alloc((size_t)MT * C * 2);
  unsigned short* hB     = (unsigned short*)alloc((size_t)MT * F * 2);

  // weights -> bf16
  f32_to_bf16_kernel<<<(C * T + 255) / 256, 256, 0, stream>>>(Wq, WqB, C * T);
  f32_to_bf16_kernel<<<(T * C + 255) / 256, 256, 0, stream>>>(Wv, WvB, T * C);
  f32_to_bf16_kernel<<<(C * F + 255) / 256, 256, 0, stream>>>(W1, W1B, C * F);
  f32_to_bf16_kernel<<<(F * C + 255) / 256, 256, 0, stream>>>(W2, W2B, F * C);

  // text prep: raw bf16 + l2-normed keys
  prep_text_kernel<<<Bb * Kk, 256, 0, stream>>>(text, textB, keytB);

  // value = text @ Wv + bv   (M=1024, N=512, K=768) -> bf16
  gemm_bf16_kernel<<<dim3(C / 128, (Bb * Kk) / 128, 1), 256, 0, stream>>>(
      textB, 0, T, WvB, 0, C, T, bvv, valueB, nullptr, nullptr, 0, 0);

  // LN1 -> bf16
  ln512_kernel<<<(unsigned)MT, 256, 0, stream>>>(visual, g1, be1, nullptr,
                                                 vislnB);

  // q = visln @ Wq + bq  (M=32768, N=768, K=512) -> bf16, then l2norm rows
  gemm_bf16_kernel<<<dim3(T / 128, (unsigned)(MT / 128), 1), 256, 0, stream>>>(
      vislnB, 0, C, WqB, 0, T, C, bq, qB, nullptr, nullptr, 0, 0);
  l2norm768_inplace_kernel<<<(unsigned)MT, 256, 0, stream>>>(qB);

  // sim + top5 + softmax -> attn bf16
  sim_softmax_kernel<<<dim3(Ntok / 64, Bb), 256, 0, stream>>>(qB, keytB,
                                                              attnB);

  // fused_pre = attn @ value + residual  (per batch, M=4096, N=512, K=128)
  gemm_bf16_kernel<<<dim3(C / 128, Ntok / 128, Bb), 256, 0, stream>>>(
      attnB, (long long)Ntok * Kk, Kk, valueB, (long long)Kk * C, C, Kk,
      nullptr, nullptr, out, visual, (long long)Ntok * C, 2);

  // LN2 in-place on d_out, also bf16 copy
  ln512_kernel<<<(unsigned)MT, 256, 0, stream>>>(out, g2, be2, out, fusedB);

  // h = gelu(fused @ W1 + b1)  (M=32768, N=2048, K=512) -> bf16
  gemm_bf16_kernel<<<dim3(F / 128, (unsigned)(MT / 128), 1), 256, 0, stream>>>(
      fusedB, 0, C, W1B, 0, F, C, b1, hB, nullptr, nullptr, 0, 1);

  // out = fused + h @ W2 + b2  (M=32768, N=512, K=2048) -> fp32 d_out
  gemm_bf16_kernel<<<dim3(C / 128, (unsigned)(MT / 128), 1), 256, 0, stream>>>(
      hB, 0, F, W2B, 0, C, F, b2, nullptr, out, out, 0, 2);
}
